// MultiHeadAttention_70935679861301
// MI455X (gfx1250) — matrix operand order repaired and checked by reading, not compiled, not run on hardware
//
#include <hip/hip_runtime.h>
#include <hip/hip_bf16.h>

typedef __bf16 bhalf;
typedef __attribute__((ext_vector_type(16))) __bf16 v16bf;
typedef __attribute__((ext_vector_type(8)))  __bf16 v8bf;
typedef __attribute__((ext_vector_type(8)))  float  v8f;

static __device__ __forceinline__ v16bf mk16(v8bf lo, v8bf hi) {
    union { v16bf v; v8bf h[2]; } u;
    u.h[0] = lo; u.h[1] = hi;
    return u.v;
}

#define WMMA_BF16(A, Bm, C) \
    __builtin_amdgcn_wmma_f32_16x16x32_bf16(false, (A), false, (Bm), (short)0, (C), false, false)

// CDNA5 async Global->LDS copy (ASYNCcnt-tracked, no VGPR round trip).
// VDST = 32-bit LDS byte address, VADDR = 64-bit global address (per-lane).
static __device__ __forceinline__ void async_copy_b128(unsigned lds_addr, const void* gaddr) {
    asm volatile("global_load_async_to_lds_b128 %0, %1, off"
                 :: "v"(lds_addr), "v"((unsigned long long)(uintptr_t)gaddr)
                 : "memory");
}
static __device__ __forceinline__ void wait_async0() {
    asm volatile("s_wait_asynccnt 0x0" ::: "memory");
}
static __device__ __forceinline__ unsigned lds_off(const void* p) {
    // Generic (flat) pointers to LDS carry the LDS byte offset in the low 32 bits
    // (flat->LDS mapping truncates to addr[31:0]).
    return (unsigned)(uintptr_t)p;
}

// Problem constants (B=16, S=1024, E=512, H=8, D=64)
constexpr int cB = 16, cS = 1024, cE = 512, cH = 8, cD = 64;
constexpr int cM = cB * cS;                       // 16384 rows
// d_out layout (floats): out | k | v
constexpr size_t K_OUT_OFF = (size_t)cB * cS * cE;   // 8388608
constexpr size_t V_OUT_OFF = 2 * K_OUT_OFF;
// workspace layout (bytes): q_bf16 | k_bf16 | vT_bf16 | ctx_bf16  (16 MB each)
constexpr size_t QB_OFF  = 0;
constexpr size_t KB_OFF  = (size_t)cB * cH * cS * cD * 2;
constexpr size_t VT_OFF  = 2 * KB_OFF;
constexpr size_t CTX_OFF = 3 * KB_OFF;

// ---------------------------------------------------------------------------
// Tiled GEMM, D = A @ W^T + bias.  A: [cM x 512] (f32 or bf16), W: [N x 512] f32.
// Block tile 128x128, K-step 32, 8 waves; each wave = 32x64 (2x4 WMMA tiles).
// EPI 0: QKV epilogue (scatter q/k/v).  EPI 1: plain f32 store (out proj).
// ---------------------------------------------------------------------------
template <int EPI, bool A_F32>
__global__ __launch_bounds__(256) void gemm_bf16_wmma(
    const void* __restrict__ Aptr, const float* __restrict__ W,
    const float* __restrict__ bias, float* __restrict__ dout,
    bhalf* __restrict__ qb, bhalf* __restrict__ kbuf, bhalf* __restrict__ vtb) {
    constexpr int Ktot = 512;
    __shared__ __align__(16) bhalf As[128][32];
    __shared__ __align__(16) bhalf Bs[128][32];

    const int tid   = threadIdx.x;
    const int m0    = blockIdx.y * 128;
    const int n0    = blockIdx.x * 128;
    const int wave  = tid >> 5;
    const int lane  = tid & 31;
    const int l15   = lane & 15;
    const int lhalf = lane >> 4;
    const int wm    = (wave >> 1) * 32;   // wave M offset inside tile
    const int wn    = (wave & 1) * 64;    // wave N offset inside tile
    const int ldrow = tid >> 1;           // 0..127
    const int ldcol = (tid & 1) * 16;     // 0 or 16

    v8f vzero = {};
    v8f acc[2][4];
#pragma unroll
    for (int i = 0; i < 2; i++)
#pragma unroll
        for (int j = 0; j < 4; j++) acc[i][j] = vzero;

    for (int k0 = 0; k0 < Ktot; k0 += 32) {
        // ---- stage A tile (convert to bf16 if needed) ----
        if constexpr (A_F32) {
            const float* src = (const float*)Aptr + (size_t)(m0 + ldrow) * Ktot + k0 + ldcol;
            v8bf t0, t1;
#pragma unroll
            for (int j = 0; j < 8; j++) { t0[j] = (bhalf)src[j]; t1[j] = (bhalf)src[8 + j]; }
            *(v8bf*)&As[ldrow][ldcol]     = t0;
            *(v8bf*)&As[ldrow][ldcol + 8] = t1;
        } else {
            const bhalf* src = (const bhalf*)Aptr + (size_t)(m0 + ldrow) * Ktot + k0 + ldcol;
            *(v8bf*)&As[ldrow][ldcol]     = ((const v8bf*)src)[0];
            *(v8bf*)&As[ldrow][ldcol + 8] = ((const v8bf*)src)[1];
        }
        // ---- stage W tile (row n of W is the B-matrix column n; K contiguous) ----
        {
            const float* src = W + (size_t)(n0 + ldrow) * Ktot + k0 + ldcol;
            v8bf t0, t1;
#pragma unroll
            for (int j = 0; j < 8; j++) { t0[j] = (bhalf)src[j]; t1[j] = (bhalf)src[8 + j]; }
            *(v8bf*)&Bs[ldrow][ldcol]     = t0;
            *(v8bf*)&Bs[ldrow][ldcol + 8] = t1;
        }
        __syncthreads();

        // ---- fragments (ISA 16-bit A/B layouts) ----
        v16bf af[2], bfr[4];
#pragma unroll
        for (int tm = 0; tm < 2; tm++) {
            const bhalf* p = &As[wm + tm * 16 + l15][0];
            af[tm] = mk16(*(const v8bf*)(p + lhalf * 8),
                          *(const v8bf*)(p + 16 + lhalf * 8));
        }
#pragma unroll
        for (int tn = 0; tn < 4; tn++) {
            const bhalf* p = &Bs[wn + tn * 16 + l15][0];
            bfr[tn] = mk16(*(const v8bf*)(p + lhalf * 8),
                           *(const v8bf*)(p + 16 + lhalf * 8));
        }
#pragma unroll
        for (int tm = 0; tm < 2; tm++)
#pragma unroll
            for (int tn = 0; tn < 4; tn++)
                acc[tm][tn] = WMMA_BF16(af[tm], bfr[tn], acc[tm][tn]);
        __syncthreads();
    }

    // ---- epilogue ----
#pragma unroll
    for (int tm = 0; tm < 2; tm++) {
#pragma unroll
        for (int tn = 0; tn < 4; tn++) {
            const int n = n0 + wn + tn * 16 + l15;
            const float bia = bias[n];
#pragma unroll
            for (int r = 0; r < 8; r++) {
                const int m = m0 + wm + tm * 16 + r + lhalf * 8;
                const float c = acc[tm][tn][r] + bia;
                if constexpr (EPI == 0) {
                    const int which = n >> 9;       // 0=q, 1=k, 2=v
                    const int nn = n & 511;
                    const int h = nn >> 6, d = nn & 63;
                    const int b = m >> 10, s = m & 1023;
                    const size_t idx = ((size_t)((b * cH + h) * cS + s)) * cD + d;
                    if (which == 0) {
                        qb[idx] = (bhalf)(c * 0.125f);   // fold 1/sqrt(D) into Q
                    } else if (which == 1) {
                        dout[K_OUT_OFF + idx] = c;       // fp32 k output
                        kbuf[idx] = (bhalf)c;
                    } else {
                        dout[V_OUT_OFF + idx] = c;       // fp32 v output
                        vtb[((size_t)((b * cH + h) * cD + d)) * cS + s] = (bhalf)c;
                    }
                } else {
                    dout[(size_t)m * cE + n] = c;
                }
            }
        }
    }
}

// ---------------------------------------------------------------------------
// Flash attention: grid (S/64, B*H), 128 threads (4 waves). Wave w owns q-rows
// [qt*64 + w*16, +16). K/V tiles are staged into LDS once per block with CDNA5
// async Global->LDS copies (shared by all 4 waves); P is transposed C-layout ->
// A-layout through per-wave LDS for the PV WMMA.
// ---------------------------------------------------------------------------
__global__ __launch_bounds__(128) void flash_attn_wmma(
    const bhalf* __restrict__ qb, const bhalf* __restrict__ kb,
    const bhalf* __restrict__ vtb, const int* __restrict__ amask,
    bhalf* __restrict__ ctx) {
    __shared__ __align__(16) bhalf Pls[4][16][64];   // per-wave P tile  (8 KB)
    __shared__ __align__(16) bhalf Ks[64][64];       // K tile [key][d]  (8 KB)
    __shared__ __align__(16) bhalf Vs[64][64];       // V tile [d][key]  (8 KB)

    const int qt    = blockIdx.x;
    const int bh    = blockIdx.y;
    const int b     = bh >> 3, h = bh & 7;
    const int tid   = threadIdx.x;
    const int wave  = tid >> 5;
    const int lane  = tid & 31;
    const int l15   = lane & 15;
    const int lhalf = lane >> 4;
    const int q0    = qt * 64 + wave * 16;   // first absolute q row of this wave

    const bhalf* qrow = qb  + (size_t)bh * cS * cD;
    const bhalf* krow = kb  + (size_t)bh * cS * cD;
    const bhalf* vrow = vtb + (size_t)bh * cD * cS;
    const int*   mrow = amask + b * cS;

    // Q A-fragments: K-dim (head dim) = 64 -> two 16x32 chunks, loaded once
    v16bf aq[2];
#pragma unroll
    for (int c = 0; c < 2; c++) {
        const bhalf* p = qrow + (size_t)(q0 + l15) * cD + c * 32;
        aq[c] = mk16(*(const v8bf*)(p + lhalf * 8),
                     *(const v8bf*)(p + 16 + lhalf * 8));
    }

    v8f vzero = {};
    v8f o[4];
#pragma unroll
    for (int tn = 0; tn < 4; tn++) o[tn] = vzero;
    float mr[8], lr[8];
#pragma unroll
    for (int r = 0; r < 8; r++) { mr[r] = -1e30f; lr[r] = 0.f; }

    const int nkb = qt + 1;   // causal: only key blocks up to the diagonal
    for (int kbk = 0; kbk < nkb; kbk++) {
        const int kb0 = kbk * 64;

        // ---- stage K/V tiles (8 KB each) via async Global->LDS, once per block ----
        // 512 16B-chunks per tile; 128 threads x 4 chunks. chunk = row*8 + c16.
#pragma unroll
        for (int i = 0; i < 4; i++) {
            const int ch  = tid + i * 128;        // 0..511
            const int row = ch >> 3;              // 0..63
            const int c16 = ch & 7;               // 16B sub-chunk within 128B row
            const bhalf* gk = krow + (size_t)(kb0 + row) * cD + c16 * 8;
            const bhalf* gv = vrow + (size_t)row * cS + kb0 + c16 * 8;
            async_copy_b128(lds_off(&Ks[row][c16 * 8]), gk);
            async_copy_b128(lds_off(&Vs[row][c16 * 8]), gv);
        }
        wait_async0();
        __syncthreads();

        // ---- S = Q @ K^T (Q pre-scaled); B-frags from LDS K tile ----
        v8f st[4];
#pragma unroll
        for (int tn = 0; tn < 4; tn++) {
            v8f a = vzero;
#pragma unroll
            for (int c = 0; c < 2; c++) {
                const bhalf* p = &Ks[tn * 16 + l15][c * 32 + lhalf * 8];
                v16bf bk = mk16(((const v8bf*)p)[0], *(const v8bf*)(p + 16));
                a = WMMA_BF16(aq[c], bk, a);
            }
            st[tn] = a;
        }

        // ---- causal + padding mask (lane owns one key column) ----
#pragma unroll
        for (int tn = 0; tn < 4; tn++) {
            const int key = kb0 + tn * 16 + l15;
            const bool dead = (mrow[key] == 0);
#pragma unroll
            for (int r = 0; r < 8; r++) {
                const int qr = q0 + r + lhalf * 8;
                if (dead || key > qr) st[tn][r] = -1e30f;
            }
        }

        // ---- online softmax (rows live in 16-lane halves of the wave) ----
#pragma unroll
        for (int r = 0; r < 8; r++) {
            float mx = fmaxf(fmaxf(st[0][r], st[1][r]), fmaxf(st[2][r], st[3][r]));
#pragma unroll
            for (int off = 1; off < 16; off <<= 1) mx = fmaxf(mx, __shfl_xor(mx, off, 32));
            const float mnew = fmaxf(mr[r], mx);
            const float rescale = __expf(mr[r] - mnew);
            mr[r] = mnew;
            float rs = 0.f;
#pragma unroll
            for (int tn = 0; tn < 4; tn++) {
                const float p = __expf(st[tn][r] - mnew);
                st[tn][r] = p;
                rs += p;
            }
#pragma unroll
            for (int off = 1; off < 16; off <<= 1) rs += __shfl_xor(rs, off, 32);
            lr[r] = lr[r] * rescale + rs;
#pragma unroll
            for (int tn = 0; tn < 4; tn++) o[tn][r] *= rescale;
        }

        // ---- P (C-layout) -> LDS row-major 16x64 bf16 ----
#pragma unroll
        for (int tn = 0; tn < 4; tn++)
#pragma unroll
            for (int r = 0; r < 8; r++)
                Pls[wave][r + lhalf * 8][tn * 16 + l15] = (bhalf)st[tn][r];
        __syncthreads();

        // ---- O += P @ V  (both operands from LDS) ----
#pragma unroll
        for (int c = 0; c < 2; c++) {
            const bhalf* pp = &Pls[wave][l15][c * 32];
            v16bf ap = mk16(*(const v8bf*)(pp + lhalf * 8),
                            *(const v8bf*)(pp + 16 + lhalf * 8));
#pragma unroll
            for (int tn = 0; tn < 4; tn++) {
                const bhalf* vp = &Vs[tn * 16 + l15][c * 32 + lhalf * 8];
                v16bf bv = mk16(((const v8bf*)vp)[0], *(const v8bf*)(vp + 16));
                o[tn] = WMMA_BF16(ap, bv, o[tn]);
            }
        }
        __syncthreads();   // protect Ks/Vs/Pls before next iteration restages
    }

    // ---- normalize and write ctx [B,S,E] (bf16, feeds out-projection GEMM) ----
#pragma unroll
    for (int r = 0; r < 8; r++) {
        const float inv = 1.0f / lr[r];
        const int s = q0 + r + lhalf * 8;
#pragma unroll
        for (int tn = 0; tn < 4; tn++) {
            const int e = h * cD + tn * 16 + l15;
            ctx[((size_t)(b * cS + s)) * cE + e] = (bhalf)(o[tn][r] * inv);
        }
    }
}

extern "C" void kernel_launch(void* const* d_in, const int* in_sizes, int n_in,
                              void* d_out, int out_size, void* d_ws, size_t ws_size,
                              hipStream_t stream) {
    (void)in_sizes; (void)n_in; (void)out_size; (void)ws_size;
    const float* x     = (const float*)d_in[0];
    const int*   amask = (const int*)d_in[1];
    const float* qkv_w = (const float*)d_in[2];
    const float* qkv_b = (const float*)d_in[3];
    const float* out_w = (const float*)d_in[4];
    const float* out_b = (const float*)d_in[5];
    float* out = (float*)d_out;

    char* ws = (char*)d_ws;
    bhalf* qb   = (bhalf*)(ws + QB_OFF);
    bhalf* kbuf = (bhalf*)(ws + KB_OFF);
    bhalf* vtb  = (bhalf*)(ws + VT_OFF);
    bhalf* ctx  = (bhalf*)(ws + CTX_OFF);

    // 1) QKV projection: writes fp32 k/v outputs + bf16 q/k/vT staging
    gemm_bf16_wmma<0, true><<<dim3(3 * cE / 128, cM / 128), 256, 0, stream>>>(
        x, qkv_w, qkv_b, out, qb, kbuf, vtb);
    // 2) Causal flash attention -> bf16 ctx
    flash_attn_wmma<<<dim3(cS / 64, cB * cH), 128, 0, stream>>>(qb, kbuf, vtb, amask, ctx);
    // 3) Output projection -> fp32 out
    gemm_bf16_wmma<1, false><<<dim3(cE / 128, cM / 128), 256, 0, stream>>>(
        ctx, out_w, out_b, out, nullptr, nullptr, nullptr);
}
